// MetaMultiHeadSelfAttention_45810121179385
// MI455X (gfx1250) — matrix operand order repaired and checked by reading, not compiled, not run on hardware
//
#include <hip/hip_runtime.h>
#include <hip/hip_bf16.h>

// ---------- types ----------
typedef __attribute__((ext_vector_type(16))) __bf16       v16bf;
typedef __attribute__((ext_vector_type(8)))  float        v8f;
typedef __attribute__((ext_vector_type(4)))  unsigned int v4u;
typedef __attribute__((ext_vector_type(4)))  float        v4f;
typedef unsigned short u16;

#define SEQ    2048
#define DMODEL 1024
#define HEADS  16
#define HDIM   64
#define NEGBIG (-1.0e30f)

union Frag { v16bf v; v4u q[2]; };

__device__ static inline u16 cvt_bf16(float f) {
    union { float f; unsigned u; } x; x.f = f;
    unsigned r = x.u + 0x7FFFu + ((x.u >> 16) & 1u);
    return (u16)(r >> 16);
}

// DPP16 ROW_XMASK lane swap within a 16-lane row: lane reads lane^XMASK.
template <int XMASK>
__device__ static inline float dpp_xor(float v) {
    int i = __builtin_bit_cast(int, v);
    int r = __builtin_amdgcn_update_dpp(i, i, 0x160 + XMASK, 0xf, 0xf, true);
    return __builtin_bit_cast(float, r);
}
__device__ static inline float red_max16(float v) {
    v = fmaxf(v, dpp_xor<1>(v));
    v = fmaxf(v, dpp_xor<2>(v));
    v = fmaxf(v, dpp_xor<4>(v));
    v = fmaxf(v, dpp_xor<8>(v));
    return v;
}
__device__ static inline float red_sum16(float v) {
    v += dpp_xor<1>(v);
    v += dpp_xor<2>(v);
    v += dpp_xor<4>(v);
    v += dpp_xor<8>(v);
    return v;
}

// A fragment (16x32 bf16, M x K): m = lane%16, element e -> k = (e/8)*16 + (lane/16)*8 + e%8
__device__ static inline Frag load_a(const u16* base, int ld, int k) {
    int lane = threadIdx.x & 31;
    int row = lane & 15, lh = lane >> 4;
    const u16* p = base + row * ld + k + lh * 8;
    Frag f;
    f.q[0] = *(const v4u*)p;
    f.q[1] = *(const v4u*)(p + 16);
    return f;
}

// B fragment (32x16 bf16, K x N): n = lane%16, element e -> k = (e/8)*16 + (lane/16)*8 + e%8
// base points at (n0, k=0) of a row-major [N x ld] matrix.
__device__ static inline Frag load_b(const u16* base, int ld, int k) {
    int lane = threadIdx.x & 31;
    int row = lane & 15, lh = lane >> 4;
    const u16* p = base + row * ld + k + lh * 8;
    Frag f;
    f.q[0] = *(const v4u*)p;
    f.q[1] = *(const v4u*)(p + 16);
    return f;
}

__device__ static inline v8f wmma_bf16(const Frag& a, const Frag& b, v8f c) {
    return __builtin_amdgcn_wmma_f32_16x16x32_bf16(
        false, a.v, false, b.v, (short)0, c, false, false);
}

// ---------- 1) f32 -> bf16 conversion ----------
__global__ __launch_bounds__(256) void cvt_kernel(const float* __restrict__ in,
                                                  u16* __restrict__ out, int n8) {
    int i = blockIdx.x * blockDim.x + threadIdx.x;
    if (i >= n8) return;
    const float* p = in + (size_t)i * 8;
    v4f f0 = *(const v4f*)p;
    v4f f1 = *(const v4f*)(p + 4);
    union { v4u q; u16 h[8]; } o;
#pragma unroll
    for (int j = 0; j < 4; ++j) { o.h[j] = cvt_bf16(f0[j]); o.h[4 + j] = cvt_bf16(f1[j]); }
    *(v4u*)(out + (size_t)i * 8) = o.q;
}

// ---------- 2/4) WMMA GEMM: C[m,n] = sum_k A[m,k] * W[n,k] ----------
// 64x64 per-wave tiles (16 WMMA : 8 fragment loads per k-step), register
// double-buffered. Block = 8 waves (2 x 4) = 128 x 256 tile.
// MODE 0: three fused weight/output pairs (QKV), out bf16 in [B,H,S,hd] layout.
// MODE 1: single weight, out f32 row-major [m,n].
template <int MODE>
__global__ __launch_bounds__(256) void gemm256(const u16* __restrict__ A,
                                               const u16* __restrict__ W0,
                                               const u16* __restrict__ W1,
                                               const u16* __restrict__ W2,
                                               void* __restrict__ o0,
                                               void* __restrict__ o1,
                                               void* __restrict__ o2) {
    int lane = threadIdx.x & 31, warp = threadIdx.x >> 5;
    int lh = lane >> 4, r16 = lane & 15;
    int wm = warp & 1, wn = warp >> 1;

    int nt   = blockIdx.x & 3;           // 4 n-tiles of 256 cover DMODEL
    int wsel = blockIdx.x >> 2;          // 0..2 in MODE 0, always 0 in MODE 1
    const u16* W = (wsel == 0) ? W0 : (wsel == 1 ? W1 : W2);
    void*    out = (wsel == 0) ? o0 : (wsel == 1 ? o1 : o2);

    int m0 = blockIdx.y * 128 + wm * 64;
    int n0 = nt * 256 + wn * 64;

    const u16* Am[4];
    const u16* Wn[4];
#pragma unroll
    for (int i = 0; i < 4; ++i) Am[i] = A + (size_t)(m0 + i * 16) * DMODEL;
#pragma unroll
    for (int j = 0; j < 4; ++j) Wn[j] = W + (size_t)(n0 + j * 16) * DMODEL;

    v8f c[4][4];
#pragma unroll
    for (int i = 0; i < 4; ++i)
#pragma unroll
        for (int j = 0; j < 4; ++j) c[i][j] = v8f{0.f,0.f,0.f,0.f,0.f,0.f,0.f,0.f};

    Frag a[4], b[4];
#pragma unroll
    for (int i = 0; i < 4; ++i) a[i] = load_a(Am[i], DMODEL, 0);
#pragma unroll
    for (int j = 0; j < 4; ++j) b[j] = load_b(Wn[j], DMODEL, 0);

    for (int k = 0; k < DMODEL; k += 32) {
        Frag an[4], bn[4];
        int kn = k + 32;
        if (kn < DMODEL) {
#pragma unroll
            for (int i = 0; i < 4; ++i) an[i] = load_a(Am[i], DMODEL, kn);
#pragma unroll
            for (int j = 0; j < 4; ++j) bn[j] = load_b(Wn[j], DMODEL, kn);
        }
#pragma unroll
        for (int i = 0; i < 4; ++i)
#pragma unroll
            for (int j = 0; j < 4; ++j)
                c[i][j] = wmma_bf16(a[i], b[j], c[i][j]);
#pragma unroll
        for (int i = 0; i < 4; ++i) a[i] = an[i];
#pragma unroll
        for (int j = 0; j < 4; ++j) b[j] = bn[j];
    }

#pragma unroll
    for (int i = 0; i < 4; ++i)
#pragma unroll
        for (int j = 0; j < 4; ++j)
#pragma unroll
            for (int r = 0; r < 8; ++r) {
                int m = m0 + i * 16 + lh * 8 + r;
                int n = n0 + j * 16 + r16;
                if (MODE == 0) {
                    int bb = m >> 11, s = m & 2047;
                    int hh = n >> 6, dh = n & 63;
                    size_t idx = (((size_t)(bb * HEADS + hh) * SEQ) + s) * HDIM + dh;
                    ((u16*)out)[idx] = cvt_bf16(c[i][j][r]);
                } else {
                    ((float*)out)[(size_t)m * DMODEL + n] = c[i][j][r];
                }
            }
}

// ---------- 3) causal flash attention ----------
// Q,K,V: [B,H,S,hd] bf16.  O: [B*S, D] bf16 (head-concat layout for O-proj GEMM).
__global__ __launch_bounds__(256) void flash_attn(const u16* __restrict__ Q,
                                                  const u16* __restrict__ K,
                                                  const u16* __restrict__ V,
                                                  u16* __restrict__ O) {
    __shared__ u16 Kt[32 * 72];        // [key][hd], padded stride 72
    __shared__ u16 Vt[64 * 40];        // [hd][key], padded stride 40 (V transposed)
    __shared__ u16 Pb[8 * 16 * 40];    // per-wave P tiles [16 q][32 k], stride 40

    int tid = threadIdx.x;
    int lane = tid & 31, warp = tid >> 5;
    int lh = lane >> 4, r16 = lane & 15;

    int tile = blockIdx.x & 15;          // S/128 = 16
    int h    = (blockIdx.x >> 4) & 15;
    int b    = blockIdx.x >> 8;
    int q0   = tile * 128;

    const u16* Qbh = Q + (size_t)(b * HEADS + h) * SEQ * HDIM;
    const u16* Kbh = K + (size_t)(b * HEADS + h) * SEQ * HDIM;
    const u16* Vbh = V + (size_t)(b * HEADS + h) * SEQ * HDIM;

    const u16* qbase = Qbh + (size_t)(q0 + warp * 16) * HDIM;
    Frag aq0 = load_a(qbase, HDIM, 0);
    Frag aq1 = load_a(qbase, HDIM, 32);

    float mrow[8], lrow[8];
    v8f o[4];
#pragma unroll
    for (int r = 0; r < 8; ++r) { mrow[r] = NEGBIG; lrow[r] = 0.f; }
#pragma unroll
    for (int f = 0; f < 4; ++f) o[f] = v8f{0.f,0.f,0.f,0.f,0.f,0.f,0.f,0.f};

    const int rK = tid >> 3, cK = (tid & 7) * 8;
    const int kend = q0 + 128;   // causal limit for this block

    // prefetch first chunk into registers
    v4u gk = *(const v4u*)&Kbh[(size_t)rK * HDIM + cK];
    v4u gv = *(const v4u*)&Vbh[(size_t)rK * HDIM + cK];

    for (int kc = 0; kc < kend; kc += 32) {
        // commit staged registers to LDS (K row-major, V transposed)
        *(v4u*)&Kt[rK * 72 + cK] = gk;
        {
            union { v4u q; u16 h8[8]; } tv; tv.q = gv;
#pragma unroll
            for (int j = 0; j < 8; ++j) Vt[(cK + j) * 40 + rK] = tv.h8[j];
        }
        __syncthreads();

        // prefetch next chunk while this one is computed
        int kn = kc + 32;
        if (kn < kend) {
            gk = *(const v4u*)&Kbh[(size_t)(kn + rK) * HDIM + cK];
            gv = *(const v4u*)&Vbh[(size_t)(kn + rK) * HDIM + cK];
        }

        // S = Q K^T : 16x32 scores per wave
        v8f c0 = v8f{0.f,0.f,0.f,0.f,0.f,0.f,0.f,0.f};
        v8f c1 = c0;
        Frag bf;
        bf = load_b(Kt, 72, 0);            c0 = wmma_bf16(aq0, bf, c0);
        bf = load_b(Kt, 72, 32);           c0 = wmma_bf16(aq1, bf, c0);
        bf = load_b(Kt + 16 * 72, 72, 0);  c1 = wmma_bf16(aq0, bf, c1);
        bf = load_b(Kt + 16 * 72, 72, 32); c1 = wmma_bf16(aq1, bf, c1);

        // scale + causal mask (mask only near the diagonal; wave-uniform branch)
        float s0a[8], s1a[8];
        bool needMask = (kc + 31) > (q0 + warp * 16);
        if (needMask) {
#pragma unroll
            for (int r = 0; r < 8; ++r) {
                int qg  = q0 + warp * 16 + lh * 8 + r;
                s0a[r] = ((kc + r16)      <= qg) ? c0[r] * 0.125f : NEGBIG;
                s1a[r] = ((kc + 16 + r16) <= qg) ? c1[r] * 0.125f : NEGBIG;
            }
        } else {
#pragma unroll
            for (int r = 0; r < 8; ++r) { s0a[r] = c0[r] * 0.125f; s1a[r] = c1[r] * 0.125f; }
        }

        // online softmax: row reductions via DPP ROW_XMASK (no LDS traffic)
        float alpha[8];
        u16* Pw = Pb + warp * 640;
#pragma unroll
        for (int r = 0; r < 8; ++r) {
            float pmax = red_max16(fmaxf(s0a[r], s1a[r]));
            float mnew = fmaxf(mrow[r], pmax);
            float a  = __expf(mrow[r] - mnew);
            float p0 = __expf(s0a[r] - mnew);
            float p1 = __expf(s1a[r] - mnew);
            float ps = red_sum16(p0 + p1);
            lrow[r] = lrow[r] * a + ps;
            mrow[r] = mnew;
            alpha[r] = a;
            Pw[(lh * 8 + r) * 40 + r16]      = cvt_bf16(p0);
            Pw[(lh * 8 + r) * 40 + 16 + r16] = cvt_bf16(p1);
        }

        // per-wave LDS round-trip to re-layout P into an A fragment
        asm volatile("s_wait_dscnt 0" ::: "memory");
        Frag ap = load_a(Pw, 40, 0);

        // O = diag(alpha) * O + P * V
#pragma unroll
        for (int f = 0; f < 4; ++f) {
#pragma unroll
            for (int r = 0; r < 8; ++r) o[f][r] *= alpha[r];
            bf = load_b(Vt + f * 16 * 40, 40, 0);
            o[f] = wmma_bf16(ap, bf, o[f]);
        }
        __syncthreads();
    }

    // epilogue: normalize and store bf16 into [B*S, D] with head-concat columns
#pragma unroll
    for (int f = 0; f < 4; ++f)
#pragma unroll
        for (int r = 0; r < 8; ++r) {
            float val = o[f][r] / lrow[r];
            int s   = q0 + warp * 16 + lh * 8 + r;
            int col = h * HDIM + f * 16 + r16;
            O[((size_t)(b * SEQ + s)) * DMODEL + col] = cvt_bf16(val);
        }
}

// ---------- launch ----------
extern "C" void kernel_launch(void* const* d_in, const int* in_sizes, int n_in,
                              void* d_out, int out_size, void* d_ws, size_t ws_size,
                              hipStream_t stream) {
    (void)in_sizes; (void)n_in; (void)out_size; (void)ws_size;

    const float* x  = (const float*)d_in[0];
    const float* wq = (const float*)d_in[1];
    const float* wk = (const float*)d_in[2];
    const float* wv = (const float*)d_in[3];
    const float* wo = (const float*)d_in[4];

    char* ws = (char*)d_ws;
    u16* xb   = (u16*)(ws + 0);               // 16 MB : x bf16 [8192,1024]
    u16* wqb  = (u16*)(ws + 16777216);        // 2 MB each
    u16* wkb  = (u16*)(ws + 18874368);
    u16* wvb  = (u16*)(ws + 20971520);
    u16* wob  = (u16*)(ws + 23068672);
    u16* Qb   = (u16*)(ws + 25165824);        // 16 MB each, [B,H,S,hd]
    u16* Kb   = (u16*)(ws + 41943040);
    u16* Vb   = (u16*)(ws + 58720256);
    u16* attb = (u16*)(ws + 75497472);        // 16 MB, [B*S, D]

    cvt_kernel<<<4096, 256, 0, stream>>>(x,  xb,  1048576);
    cvt_kernel<<<512,  256, 0, stream>>>(wq, wqb, 131072);
    cvt_kernel<<<512,  256, 0, stream>>>(wk, wkb, 131072);
    cvt_kernel<<<512,  256, 0, stream>>>(wv, wvb, 131072);
    cvt_kernel<<<512,  256, 0, stream>>>(wo, wob, 131072);

    // fused Q/K/V projections: grid.x = 3 weights x 4 n-tiles
    gemm256<0><<<dim3(12, 64), 256, 0, stream>>>(xb, wqb, wkb, wvb,
                                                 (void*)Qb, (void*)Kb, (void*)Vb);

    flash_attn<<<4 * HEADS * (SEQ / 128), 256, 0, stream>>>(Qb, Kb, Vb, attb);

    // output projection -> f32 d_out
    gemm256<1><<<dim3(4, 64), 256, 0, stream>>>(attb, wob, wob, wob,
                                                d_out, d_out, d_out);
}
